// Encoder_74028056314445
// MI455X (gfx1250) — compile-verified, matrix operand order checked by reading, not run on hardware
//
#include <hip/hip_runtime.h>
#include <hip/hip_bf16.h>
#include <math.h>

// ---------------------------------------------------------------------------
// Types for CDNA5 WMMA (wave32): V_WMMA_F32_16X16X32_BF16
// ---------------------------------------------------------------------------
typedef __bf16 bf16;
typedef __attribute__((ext_vector_type(16))) __bf16 v16bf;
typedef __attribute__((ext_vector_type(8)))  float  v8f;
typedef __attribute__((ext_vector_type(2)))  float  f32x2;   // native vec for NT load

#define TOPICS  64
#define WORDS   10000
#define BATCH_N 256
#define FAN     10

// Fragment K-offset for (v, hi):  ks = 2v + (v>=4 ? 8 : 0) + (hi ? 8 : 0)
// (CDNA5 16-bit A/B VGPR layout, ISA 7.12.2)

// ---------------------------------------------------------------------------
// One-time pack of B = W[K,N] (f32, row-major) into bf16 WMMA fragment order:
// Bp[kb][ntile][lane][0..15]  (16 bf16 = 32B contiguous per lane),
// zero-filled past K so the GEMM needs no bounds checks on B.
// grid = KB*(N/16) tiles, block = 32.
// ---------------------------------------------------------------------------
__global__ void pack_b_wmma(const float* __restrict__ B, bf16* __restrict__ Bp,
                            int N, int K)
{
    const int tile = blockIdx.x;
    const int NT   = N >> 4;
    const int kb   = tile / NT;
    const int nt   = tile % NT;
    const int lane = threadIdx.x;
    const int lr   = lane & 15;
    const int hi   = lane >> 4;

    bf16* dst = Bp + ((size_t)tile * 32 + lane) * 16;
    #pragma unroll
    for (int v = 0; v < 8; ++v) {
        const int ks = 2 * v + (v >= 4 ? 8 : 0) + (hi ? 8 : 0);
        const int kg = kb * 32 + ks;
        const float b0 = (kg     < K) ? B[(size_t)kg       * N + nt * 16 + lr] : 0.0f;
        const float b1 = (kg + 1 < K) ? B[(size_t)(kg + 1) * N + nt * 16 + lr] : 0.0f;
        dst[2 * v]     = (bf16)b0;
        dst[2 * v + 1] = (bf16)b1;
    }
}

// ---------------------------------------------------------------------------
// GEMM: C[M,N] = A[M,K] @ W[K,N] + bias[N]; A f32 (converted to bf16 in LDS,
// fragment order, double-buffered), B pre-packed bf16, f32 WMMA accumulate.
// grid.x = M/16.  block = NTHR = (N/16)*32 threads (one wave per N-tile).
// Staging is straight-line (no bounds checks, no exec-mask regions) in the
// steady state; only the final K-block (K%32 tail) is staged with a guard.
// ---------------------------------------------------------------------------
template<int NTHR>
__global__ void __launch_bounds__(NTHR)
gemm_bias_wmma(const float* __restrict__ A,
               const bf16*  __restrict__ Bp,
               const float* __restrict__ bias,
               float* __restrict__ C,
               int M, int N, int K)
{
    static_assert(NTHR == 128 || NTHR == 256, "block must be 128 or 256");
    // 2 buffers x 32 lanes x 8 dwords (16 bf16 per lane, fragment order)
    __shared__ __align__(16) unsigned As[2][256];

    const int m0    = blockIdx.x * 16;
    const int ntile = threadIdx.x >> 5;
    const int lane  = threadIdx.x & 31;
    const int lr    = lane & 15;
    const int hi    = lane >> 4;
    const int KB    = (K + 31) >> 5;
    const int NT    = N >> 4;

    // Stage one element pair of the 16x32 A tile (f32 -> bf16) into LDS in
    // fragment order. Element s: A[m0 + (s>>4)][kb*32 + 2*(s&15) .. +1].
    auto stage_elem = [&](int buf, int kb, int s) {          // unguarded
        const int r  = s >> 4;
        const int kk = (s & 15) << 1;
        const f32x2 f = __builtin_nontemporal_load(
            (const f32x2*)(A + (size_t)(m0 + r) * K + kb * 32 + kk));
        const int hi2 = (kk >> 3) & 1;
        const int v   = ((kk & 16) >> 2) + ((kk & 6) >> 1);
        union { bf16 h[2]; unsigned u; } p;
        p.h[0] = (bf16)f.x;
        p.h[1] = (bf16)f.y;
        As[buf][(r + (hi2 << 4)) * 8 + v] = p.u;
    };
    auto stage_elem_t = [&](int buf, int kb, int s) {        // guarded (tail)
        const int r  = s >> 4;
        const int kk = (s & 15) << 1;
        const int kg = kb * 32 + kk;
        f32x2 f = (f32x2)(0.0f);
        if (kg < K)                               // K even, kk even -> pair safe
            f = __builtin_nontemporal_load(
                    (const f32x2*)(A + (size_t)(m0 + r) * K + kg));
        const int hi2 = (kk >> 3) & 1;
        const int v   = ((kk & 16) >> 2) + ((kk & 6) >> 1);
        union { bf16 h[2]; unsigned u; } p;
        p.h[0] = (bf16)f.x;
        p.h[1] = (bf16)f.y;
        As[buf][(r + (hi2 << 4)) * 8 + v] = p.u;
    };
    auto stage_full = [&](int buf, int kb) {   // straight-line, zero compares
        if constexpr (NTHR >= 256) {
            stage_elem(buf, kb, (int)threadIdx.x);
        } else {
            #pragma unroll
            for (int i = 0; i < 256 / NTHR; ++i)
                stage_elem(buf, kb, (int)threadIdx.x + i * NTHR);
        }
    };
    auto stage_tail = [&](int buf, int kb) {
        if constexpr (NTHR >= 256) {
            stage_elem_t(buf, kb, (int)threadIdx.x);
        } else {
            #pragma unroll
            for (int i = 0; i < 256 / NTHR; ++i)
                stage_elem_t(buf, kb, (int)threadIdx.x + i * NTHR);
        }
    };

    v8f acc = {};
    if (KB > 1) stage_full(0, 0); else stage_tail(0, 0);
    __syncthreads();

    // Per-wave packed-B pointer; advances NT*64 uint4 per K-block.
    const uint4* bptr = (const uint4*)Bp + ((size_t)ntile * 32 + lane) * 2;
    const size_t bstep = (size_t)NT * 64;

    union { uint4 q[2]; v16bf v; } af, bf_;
    int kb = 0;

    // steady state: no bounds checks, no exec-mask regions
    for (; kb + 2 < KB; ++kb) {
        const int buf = kb & 1;
        stage_full(buf ^ 1, kb + 1);
        const uint4* asrc = (const uint4*)&As[buf][0];
        af.q[0] = asrc[lane * 2];                    // 2 x ds_load_b128
        af.q[1] = asrc[lane * 2 + 1];
        bf_.q[0] = bptr[0];                          // 2 x global_load_b128
        bf_.q[1] = bptr[1];
        bptr += bstep;
        __syncthreads();
        acc = __builtin_amdgcn_wmma_f32_16x16x32_bf16(
                  false, af.v, false, bf_.v, (short)0, acc, false, false);
    }
    // second-to-last block: stage the (possibly partial) tail block
    if (kb + 1 < KB) {
        const int buf = kb & 1;
        stage_tail(buf ^ 1, kb + 1);
        const uint4* asrc = (const uint4*)&As[buf][0];
        af.q[0] = asrc[lane * 2];
        af.q[1] = asrc[lane * 2 + 1];
        bf_.q[0] = bptr[0];
        bf_.q[1] = bptr[1];
        bptr += bstep;
        __syncthreads();
        acc = __builtin_amdgcn_wmma_f32_16x16x32_bf16(
                  false, af.v, false, bf_.v, (short)0, acc, false, false);
        ++kb;
    }
    // last block: no staging, no barrier needed after
    {
        const int buf = kb & 1;
        const uint4* asrc = (const uint4*)&As[buf][0];
        af.q[0] = asrc[lane * 2];
        af.q[1] = asrc[lane * 2 + 1];
        bf_.q[0] = bptr[0];
        bf_.q[1] = bptr[1];
        acc = __builtin_amdgcn_wmma_f32_16x16x32_bf16(
                  false, af.v, false, bf_.v, (short)0, acc, false, false);
    }

    // D layout: VGPR r -> M = r + 8*hi, N = lane&15
    const int n = ntile * 16 + lr;
    const float bn = bias ? bias[n] : 0.0f;
    #pragma unroll
    for (int r = 0; r < 8; ++r) {
        const int m = r + (hi ? 8 : 0);
        C[(size_t)(m0 + m) * N + n] = acc[r] + bn;
    }
}

// ---------------------------------------------------------------------------
// Attention + aggregation: one wave per output row (wave32).
// att_j = softmax_j( sigmoid( Hs[i].aw[0:d] + Hn[i*10+j].aw[d:2d] + ab ) )
// out   = act( 0.5*Hs[i] + 0.5*sum_j att_j * Hn[i*10+j] )
// ---------------------------------------------------------------------------
__global__ void attn_agg(const float* __restrict__ Hs,
                         const float* __restrict__ Hn,
                         const float* __restrict__ aw,
                         const float* __restrict__ ab,
                         float* __restrict__ Out,
                         float* __restrict__ AttOut,
                         int N, int d, int act_tanh)
{
    const int wave = threadIdx.x >> 5;
    const int lane = threadIdx.x & 31;
    const int row  = blockIdx.x * (blockDim.x >> 5) + wave;
    if (row >= N) return;

    const int cpl = d >> 5;                 // cols per lane (4 for d=128, 2 for d=64)
    const float* hs = Hs + (size_t)row * d;

    float sself = 0.0f;
    for (int t = 0; t < cpl; ++t) {
        const int c = lane + (t << 5);
        sself += hs[c] * aw[c];
    }
    #pragma unroll
    for (int m = 16; m; m >>= 1) sself += __shfl_xor(sself, m, 32);

    float attv[FAN];
    #pragma unroll
    for (int j = 0; j < FAN; ++j) {
        const float* hn = Hn + (size_t)(row * FAN + j) * d;
        float s = 0.0f;
        for (int t = 0; t < cpl; ++t) {
            const int c = lane + (t << 5);
            s += hn[c] * aw[d + c];
        }
        #pragma unroll
        for (int m = 16; m; m >>= 1) s += __shfl_xor(s, m, 32);
        const float z = sself + s + ab[0];
        attv[j] = 1.0f / (1.0f + expf(-z));     // sigmoid
    }
    // softmax over the sigmoid scores
    float mx = attv[0];
    #pragma unroll
    for (int j = 1; j < FAN; ++j) mx = fmaxf(mx, attv[j]);
    float ssum = 0.0f;
    #pragma unroll
    for (int j = 0; j < FAN; ++j) { attv[j] = expf(attv[j] - mx); ssum += attv[j]; }
    const float inv = 1.0f / ssum;
    #pragma unroll
    for (int j = 0; j < FAN; ++j) attv[j] *= inv;

    for (int t = 0; t < cpl; ++t) {
        const int c = lane + (t << 5);
        float agg = 0.0f;
        #pragma unroll
        for (int j = 0; j < FAN; ++j)
            agg += attv[j] * Hn[(size_t)(row * FAN + j) * d + c];
        float o = 0.5f * hs[c] + 0.5f * agg;
        if (act_tanh) o = tanhf(o);
        Out[(size_t)row * d + c] = o;
    }
    if (AttOut) {
        #pragma unroll
        for (int j = 0; j < FAN; ++j)
            if (lane == j) AttOut[row * FAN + j] = attv[j];
    }
}

// ---------------------------------------------------------------------------
// BatchNorm (training: batch stats, biased var, gamma=1 beta=0) + softplus
// ---------------------------------------------------------------------------
__global__ void batchnorm_softplus(const float* __restrict__ X,
                                   float* __restrict__ MeanOut,
                                   float* __restrict__ Alpha,
                                   int Nrows, int D)
{
    const int c = threadIdx.x;
    if (c >= D) return;
    float s = 0.0f, s2 = 0.0f;
    for (int r = 0; r < Nrows; ++r) {
        const float v = X[(size_t)r * D + c];
        s += v; s2 += v * v;
    }
    const float mu  = s / (float)Nrows;
    const float var = s2 / (float)Nrows - mu * mu;
    const float rstd = rsqrtf(var + 1e-5f);
    for (int r = 0; r < Nrows; ++r) {
        const float m = (X[(size_t)r * D + c] - mu) * rstd;
        MeanOut[(size_t)r * D + c] = m;
        Alpha[(size_t)r * D + c]  = fmaxf(1e-12f, log1pf(expf(m)));
    }
}

// ---------------------------------------------------------------------------
// Counter-based deterministic RNG helpers
// ---------------------------------------------------------------------------
__device__ inline unsigned pcg_hash(unsigned x) {
    x = x * 747796405u + 2891336453u;
    unsigned w = ((x >> ((x >> 28u) + 4u)) ^ x) * 277803737u;
    return (w >> 22u) ^ w;
}
__device__ inline float urand01(unsigned s) {   // in (0,1), never 0
    return ((float)((pcg_hash(s) >> 8) + 0.5f)) * (1.0f / 16777216.0f);
}

// ---------------------------------------------------------------------------
// Gamma reparameterization + row-normalize -> doc_topic_dist.
// gamma_tmp in the reference collapses algebraically to the raw Gamma(aB)
// draw; sample Gamma(alpha+10) (Marsaglia-Tsang) and apply 10 boost uniforms.
// ---------------------------------------------------------------------------
__global__ void gamma_dist(const float* __restrict__ Alpha,
                           float* __restrict__ Dist,
                           int Nrows, int D)
{
    const int wave = threadIdx.x >> 5;
    const int lane = threadIdx.x & 31;
    const int row  = blockIdx.x * (blockDim.x >> 5) + wave;
    if (row >= Nrows) return;

    float g2[2];
    #pragma unroll
    for (int t = 0; t < 2; ++t) {
        const int c = lane + (t << 5);
        const float a  = Alpha[(size_t)row * D + c];
        const float aB = a + 10.0f;
        const unsigned base = (unsigned)(row * D + c) * 97u + 0x9E3779B9u;

        const float dd = aB - (1.0f / 3.0f);
        const float cc = rsqrtf(9.0f * dd);
        float g = dd;                         // fallback if no accept (p ~ 0)
        for (int it2 = 0; it2 < 16; ++it2) {
            const float u1 = urand01(base + 4u * it2 + 0u);
            const float u2 = urand01(base + 4u * it2 + 1u);
            const float u3 = urand01(base + 4u * it2 + 2u);
            const float x  = sqrtf(-2.0f * logf(u1)) * cosf(6.28318530718f * u2);
            float v = 1.0f + cc * x;
            if (v <= 0.0f) continue;
            v = v * v * v;
            if (logf(u3) < 0.5f * x * x + dd - dd * v + dd * logf(v)) {
                g = dd * v;
                break;
            }
        }
        float prod = 1.0f;
        #pragma unroll
        for (int i = 0; i < 10; ++i) {
            const float u = urand01(base + 64u + (unsigned)i);
            prod *= powf(u, 1.0f / (a + (float)i)) + 1e-10f;
        }
        g2[t] = g * prod;
    }
    float s = g2[0] + g2[1];
    #pragma unroll
    for (int m = 16; m; m >>= 1) s += __shfl_xor(s, m, 32);
    const float inv = 1.0f / s;
    #pragma unroll
    for (int t = 0; t < 2; ++t)
        Dist[(size_t)row * D + lane + (t << 5)] = g2[t] * inv;
}

// ---------------------------------------------------------------------------
// Dirichlet KL(alpha || 0.1) -> mean scalar
// ---------------------------------------------------------------------------
__device__ inline float digammaf_(float x) {
    float r = 0.0f;
    while (x < 6.0f) { r -= 1.0f / x; x += 1.0f; }
    const float i1 = 1.0f / x, i2 = i1 * i1;
    return r + logf(x) - 0.5f * i1
             - i2 * (1.0f / 12.0f - i2 * (1.0f / 120.0f - i2 * (1.0f / 252.0f)));
}

__global__ void kld_kernel(const float* __restrict__ Alpha,
                           float* __restrict__ OutScalar,
                           int Nrows, int D)
{
    __shared__ float red[256];
    const int r = threadIdx.x;
    float kl = 0.0f;
    if (r < Nrows) {
        float asum = 0.0f;
        for (int c = 0; c < D; ++c) asum += Alpha[(size_t)r * D + c];
        const float dga = digammaf_(asum);
        float t2 = 0.0f, t3 = 0.0f;
        for (int c = 0; c < D; ++c) {
            const float a = Alpha[(size_t)r * D + c];
            t2 += lgammaf(a);
            t3 += (a - 0.1f) * (digammaf_(a) - dga);
        }
        kl = lgammaf(asum) - lgammaf(0.1f * (float)D)
           - t2 + (float)D * lgammaf(0.1f) + t3;
    }
    red[threadIdx.x] = kl;
    __syncthreads();
    for (int s = 128; s; s >>= 1) {
        if (threadIdx.x < s) red[threadIdx.x] += red[threadIdx.x + s];
        __syncthreads();
    }
    if (threadIdx.x == 0) OutScalar[0] = red[0] / (float)Nrows;
}

// ---------------------------------------------------------------------------
// Launch: pack W0/W1 -> 3 big GEMMs (bf16 WMMA) -> attn layer0 (x2, tanh)
// -> 2 small GEMMs -> attn layer1 (ident, emits att) -> BN+softplus ->
// gamma sample -> KL
// ---------------------------------------------------------------------------
extern "C" void kernel_launch(void* const* d_in, const int* in_sizes, int n_in,
                              void* d_out, int out_size, void* d_ws, size_t ws_size,
                              hipStream_t stream)
{
    const float* feat0 = (const float*)d_in[0];
    const float* feat1 = (const float*)d_in[1];
    const float* feat2 = (const float*)d_in[2];
    const float* W0    = (const float*)d_in[3];
    const float* b0    = (const float*)d_in[4];
    const float* a0w   = (const float*)d_in[5];
    const float* a0b   = (const float*)d_in[6];
    const float* W1    = (const float*)d_in[7];
    const float* b1    = (const float*)d_in[8];
    const float* a1w   = (const float*)d_in[9];
    const float* a1b   = (const float*)d_in[10];
    (void)in_sizes; (void)n_in; (void)out_size; (void)ws_size;

    float* out = (float*)d_out;
    float* out_dist = out;                        // [256*64]
    float* out_mean = out + BATCH_N * TOPICS;     // [256*64]
    float* out_kld  = out + 2 * BATCH_N * TOPICS; // [1]
    float* out_att  = out_kld + 1;                // [256*10]

    float* ws    = (float*)d_ws;
    float* H0    = ws;  ws += BATCH_N * 128;              // 32768
    float* H1    = ws;  ws += BATCH_N * FAN * 128;        // 327680
    float* H2    = ws;  ws += BATCH_N * FAN * FAN * 128;  // 3276800
    float* e0    = ws;  ws += BATCH_N * 128;
    float* e1    = ws;  ws += BATCH_N * FAN * 128;
    float* G0    = ws;  ws += BATCH_N * TOPICS;
    float* G1    = ws;  ws += BATCH_N * FAN * TOPICS;
    float* demb  = ws;  ws += BATCH_N * TOPICS;
    float* alpha = ws;  ws += BATCH_N * TOPICS;
    bf16*  Bp0   = (bf16*)ws;   // packed W0: 313*8 tiles * 512 bf16 = 2.45 MB
    const int KB0 = (WORDS + 31) / 32, NT0 = 128 / 16;    // 313, 8
    bf16*  Bp1   = Bp0 + (size_t)KB0 * NT0 * 512;         // packed W1: 16 KB
    const int KB1 = 128 / 32, NT1 = TOPICS / 16;          // 4, 4

    const int M0 = BATCH_N, M1 = BATCH_N * FAN, M2 = BATCH_N * FAN * FAN;

    // --- pack weights into WMMA fragment order (zero-filled K tail) ---
    pack_b_wmma<<<dim3(KB0 * NT0), dim3(32), 0, stream>>>(W0, Bp0, 128, WORDS);
    pack_b_wmma<<<dim3(KB1 * NT1), dim3(32), 0, stream>>>(W1, Bp1, TOPICS, 128);

    // --- layer-0 linear (shared W0), computed once per hop ---
    gemm_bias_wmma<256><<<dim3(M0 / 16), dim3(256), 0, stream>>>(feat0, Bp0, b0, H0, M0, 128, WORDS);
    gemm_bias_wmma<256><<<dim3(M1 / 16), dim3(256), 0, stream>>>(feat1, Bp0, b0, H1, M1, 128, WORDS);
    gemm_bias_wmma<256><<<dim3(M2 / 16), dim3(256), 0, stream>>>(feat2, Bp0, b0, H2, M2, 128, WORDS);

    // --- layer-0 attention + aggregation (tanh) ---
    attn_agg<<<dim3(M0 / 8), dim3(256), 0, stream>>>(H0, H1, a0w, a0b, e0, nullptr, M0, 128, 1);
    attn_agg<<<dim3(M1 / 8), dim3(256), 0, stream>>>(H1, H2, a0w, a0b, e1, nullptr, M1, 128, 1);

    // --- layer-1 linear (W1: 128 -> 64) ---
    gemm_bias_wmma<128><<<dim3(M0 / 16), dim3(128), 0, stream>>>(e0, Bp1, b1, G0, M0, TOPICS, 128);
    gemm_bias_wmma<128><<<dim3(M1 / 16), dim3(128), 0, stream>>>(e1, Bp1, b1, G1, M1, TOPICS, 128);

    // --- layer-1 attention (identity act), att goes to output ---
    attn_agg<<<dim3(M0 / 8), dim3(256), 0, stream>>>(G0, G1, a1w, a1b, demb, out_att, M0, TOPICS, 0);

    // --- batchnorm -> mean output, softplus -> alpha ---
    batchnorm_softplus<<<dim3(1), dim3(TOPICS), 0, stream>>>(demb, out_mean, alpha, BATCH_N, TOPICS);

    // --- gamma reparameterization + normalize -> doc_topic_dist ---
    gamma_dist<<<dim3(BATCH_N / 8), dim3(256), 0, stream>>>(alpha, out_dist, BATCH_N, TOPICS);

    // --- Dirichlet KL mean -> scalar ---
    kld_kernel<<<dim3(1), dim3(256), 0, stream>>>(alpha, out_kld, BATCH_N, TOPICS);
}
